// DiffusionGraphConv_10273561772732
// MI455X (gfx1250) — matrix operand order repaired and checked by reading, not compiled, not run on hardware
//
#include <hip/hip_runtime.h>

// ---------------------------------------------------------------------------
// DiffusionGraphConv on gfx1250 (MI455X), f16 WMMA pipeline.
//   N=1024 nodes, B=64 batch, F=128 features, COLS=B*F=8192, OUT=128, M=5.
// Everything is cast to f16 once, all GEMMs run on v_wmma_f32_16x16x32_f16
// with f32 accumulation, final output is f32.
// ---------------------------------------------------------------------------

typedef __attribute__((ext_vector_type(16))) _Float16     v16h;
typedef __attribute__((ext_vector_type(8)))  float        v8f;
typedef __attribute__((ext_vector_type(4)))  unsigned int u32x4;
typedef __attribute__((ext_vector_type(4)))  float        f32x4;

#define N_NODES 1024
#define COLS    8192      // BATCH * F
#define FDIM    128
#define ODIM    128

union HFrag { v16h h; u32x4 q[2]; };
union H8    { u32x4 q; _Float16 h[8]; };

// A-fragment (16x32 f16, M x K), row-major source, lane r = M row,
// hb selects the contiguous K half (k = 16*hb + s): two contiguous 16B chunks per lane.
static __device__ __forceinline__ v16h ld_a_row(const _Float16* rowBase, int kk, int hb) {
  HFrag u;
  u.q[0] = *reinterpret_cast<const u32x4*>(rowBase + kk + hb * 16);
  u.q[1] = *reinterpret_cast<const u32x4*>(rowBase + kk + hb * 16 + 8);
  return u.h;
}

// B-fragment (32x16 f16, K x N), column-major source (colBase already
// includes the per-lane column offset and hb*16): two contiguous 16B chunks.
static __device__ __forceinline__ v16h ld_b_col(const _Float16* colBase, int kk) {
  HFrag u;
  u.q[0] = *reinterpret_cast<const u32x4*>(colBase + kk);
  u.q[1] = *reinterpret_cast<const u32x4*>(colBase + kk + 8);
  return u.h;
}

// ---------------------------------------------------------------------------
// Pack x0 = concat(inputs, state) -> f16, in BOTH layouts:
//   X [n][b*128+f] (row-major) and Xt[b*128+f][n] (transposed),
// using an LDS-tiled transpose so both global writes are coalesced.
// grid (256, 32), block (32, 8)
// ---------------------------------------------------------------------------
__global__ __launch_bounds__(256) void pack_x0_kernel(
    const float* __restrict__ inp, const float* __restrict__ st,
    _Float16* __restrict__ X, _Float16* __restrict__ Xt) {
  __shared__ _Float16 tile[32][33];
  const int n0 = blockIdx.y * 32;
  const int c0 = blockIdx.x * 32;
  const int tx = threadIdx.x;   // 0..31
  const int ty = threadIdx.y;   // 0..7
#pragma unroll
  for (int j = 0; j < 4; ++j) {
    const int n   = n0 + ty + j * 8;
    const int col = c0 + tx;
    const int b = col >> 7, f = col & 127;
    const float v = (f < 64) ? inp[((size_t)b << 16) + (n << 6) + f]
                             : st [((size_t)b << 16) + (n << 6) + (f - 64)];
    const _Float16 h = (_Float16)v;
    X[(size_t)n * COLS + col] = h;
    tile[ty + j * 8][tx] = h;
  }
  __syncthreads();
#pragma unroll
  for (int j = 0; j < 4; ++j) {
    const int cc = ty + j * 8;
    Xt[(size_t)(c0 + cc) * N_NODES + n0 + tx] = tile[tx][cc];
  }
}

// supports f32 [2][1024][1024] -> f16, same order.
__global__ __launch_bounds__(256) void convert_a_kernel(
    const float* __restrict__ S, _Float16* __restrict__ Ah) {
  const size_t idx = (size_t)blockIdx.x * 256 + threadIdx.x;
  Ah[idx] = (_Float16)S[idx];
}

// Gather Wt[k][o][f] = weight[(f*5 + k)*128 + o] as f16 (row-major in f).
__global__ __launch_bounds__(256) void gather_w_kernel(
    const float* __restrict__ W, _Float16* __restrict__ Wt) {
  const int idx = blockIdx.x * 256 + threadIdx.x;  // over 5*128*128
  const int k = idx >> 14;
  const int o = (idx >> 7) & 127;
  const int f = idx & 127;
  Wt[idx] = (_Float16)W[(f * 5 + k) * ODIM + o];
}

// ---------------------------------------------------------------------------
// Diffusion GEMM:  Y[n][col] = alpha * sum_m A[n][m] * X[m][col]  (- Z if has_z)
//   A  : f16 row-major [1024x1024]
//   Xt : f16 transposed [col][m]  (column-major B operand -> contiguous K)
//   Zt : f16 transposed [col][n]  (x0 for the Chebyshev step)
//   Y  : row-major out (always, via per-wave LDS transpose -> b128 stores),
//   Yt : transposed out (if write_t, packed 16B per lane).
// Block: 256 threads = 8 waves, wave tile 32(M) x 64(N), block tile 64 x 256.
// grid (8192/256=32, 1024/64=16)
// ---------------------------------------------------------------------------
__global__ __launch_bounds__(256) void diff_gemm_kernel(
    const _Float16* __restrict__ A, const _Float16* __restrict__ Xt,
    const _Float16* __restrict__ Zt, _Float16* __restrict__ Y,
    _Float16* __restrict__ Yt, float alpha, int has_z, int write_t) {
  // Per-wave output staging tile: 32 rows x 64 cols, padded to a 72-element
  // (144 B) row pitch so 16B vectors stay aligned. 8 waves -> 36 KB LDS.
  __shared__ _Float16 ytile[8][32][72];

  const int lane = threadIdx.x & 31;
  const int w    = threadIdx.x >> 5;
  const int wm   = w & 1;
  const int wn   = w >> 1;
  const int m0   = blockIdx.y * 64 + wm * 32;
  const int n0   = blockIdx.x * 256 + wn * 64;
  const int r    = lane & 15;
  const int hb   = lane >> 4;

  v8f acc[2][4] = {};

  const _Float16* aRow0 = A + (size_t)(m0 + r) * N_NODES;
  const _Float16* aRow1 = A + (size_t)(m0 + 16 + r) * N_NODES;
  const _Float16* bCol[4];
#pragma unroll
  for (int j = 0; j < 4; ++j)
    bCol[j] = Xt + (size_t)(n0 + j * 16 + r) * N_NODES + hb * 16;

  for (int kk = 0; kk < N_NODES; kk += 32) {
    __builtin_prefetch(aRow0 + kk + 64, 0, 3);    // -> global_prefetch_b8
    __builtin_prefetch(bCol[0] + kk + 64, 0, 3);  // streaming X panel
    const v16h a0 = ld_a_row(aRow0, kk, hb);
    const v16h a1 = ld_a_row(aRow1, kk, hb);
#pragma unroll
    for (int j = 0; j < 4; ++j) {
      const v16h b = ld_b_col(bCol[j], kk);
      acc[0][j] = __builtin_amdgcn_wmma_f32_16x16x32_f16(
          false, a0, false, b, (short)0, acc[0][j], false, false);
      acc[1][j] = __builtin_amdgcn_wmma_f32_16x16x32_f16(
          false, a1, false, b, (short)0, acc[1][j], false, false);
    }
  }

  // Epilogue: y = alpha*acc (- z).  C layout: c[t] -> row m0+i*16+8*hb+t,
  // col n0+j*16+r.  Transposed store is already contiguous; the row-major
  // store goes through the per-wave LDS tile to become b128 stores.
  _Float16 (*yt)[72] = ytile[w];
#pragma unroll
  for (int i = 0; i < 2; ++i) {
#pragma unroll
    for (int j = 0; j < 4; ++j) {
      const int rrel = i * 16 + hb * 8;       // row within wave tile
      const int crel = j * 16 + r;            // col within wave tile
      const int col  = n0 + crel;
      H8 z; H8 hout;
      if (has_z)
        z.q = *reinterpret_cast<const u32x4*>(Zt + (size_t)col * N_NODES + m0 + rrel);
#pragma unroll
      for (int t = 0; t < 8; ++t) {
        float y = alpha * acc[i][j][t];
        if (has_z) y -= (float)z.h[t];
        hout.h[t] = (_Float16)y;
      }
      if (write_t)
        *reinterpret_cast<u32x4*>(Yt + (size_t)col * N_NODES + m0 + rrel) = hout.q;
#pragma unroll
      for (int t = 0; t < 8; ++t)
        yt[rrel + t][crel] = hout.h[t];       // ds_store_b16 (LDS scatter)
    }
  }

  // Wave-local readback (intra-wave DS ordering; no block barrier needed):
  // lane owns global row m0+lane, writes 64 cols as 8x b128.
  _Float16* ydst = Y + (size_t)(m0 + lane) * COLS + n0;
#pragma unroll
  for (int c = 0; c < 8; ++c) {
    *reinterpret_cast<u32x4*>(ydst + c * 8) =
        *reinterpret_cast<const u32x4*>(&yt[lane][c * 8]);   // ds_load_b128
  }
}

// ---------------------------------------------------------------------------
// Output GEMM (accumulated over the 5 diffusion matrices), computed as
//   out[row][o] += sum_f Wt[k][o][f] * Xk[row][f],  row = b*1024 + n,
// i.e. out^T tiles: A = Wt[k] (o x f, row-major), B = Xk (f x row).
// Block: 256 threads = 8 waves; wave w owns o in [16w,16w+16); 64 rows/block.
// grid (65536/64 = 1024)
// ---------------------------------------------------------------------------
__global__ __launch_bounds__(256) void out_gemm_kernel(
    const _Float16* __restrict__ Wt, const _Float16* __restrict__ X,
    const float* __restrict__ bias, float* __restrict__ out, int init) {
  const int lane = threadIdx.x & 31;
  const int w    = threadIdx.x >> 5;
  const int o0   = w * 16;
  const int r0   = blockIdx.x * 64;
  const int r    = lane & 15;
  const int hb   = lane >> 4;

  v8f acc[4] = {};

  const _Float16* aRow = Wt + (size_t)(o0 + r) * FDIM;
  const _Float16* bRow[4];
#pragma unroll
  for (int j = 0; j < 4; ++j) {
    const int rowIdx = r0 + j * 16 + r;
    const int b = rowIdx >> 10;
    const int n = rowIdx & 1023;
    bRow[j] = X + (size_t)n * COLS + b * FDIM + hb * 16;
  }

#pragma unroll
  for (int kk = 0; kk < FDIM; kk += 32) {
    const v16h a = ld_a_row(aRow, kk, hb);
#pragma unroll
    for (int j = 0; j < 4; ++j) {
      const v16h b = ld_b_col(bRow[j], kk);
      acc[j] = __builtin_amdgcn_wmma_f32_16x16x32_f16(
          false, a, false, b, (short)0, acc[j], false, false);
    }
  }

#pragma unroll
  for (int j = 0; j < 4; ++j) {
    const int rowIdx = r0 + j * 16 + r;
    float* p = out + (size_t)rowIdx * ODIM + o0 + hb * 8;
    f32x4* p4 = reinterpret_cast<f32x4*>(p);
    f32x4 v0, v1;
    if (init) {
#pragma unroll
      for (int t = 0; t < 4; ++t) {
        v0[t] = acc[j][t]     + bias[o0 + hb * 8 + t];
        v1[t] = acc[j][4 + t] + bias[o0 + hb * 8 + 4 + t];
      }
    } else {
      v0 = p4[0]; v1 = p4[1];
#pragma unroll
      for (int t = 0; t < 4; ++t) { v0[t] += acc[j][t]; v1[t] += acc[j][4 + t]; }
    }
    p4[0] = v0; p4[1] = v1;
  }
}

// ---------------------------------------------------------------------------
// Workspace layout (f16 elements):
//   X0  row-major   1024*8192            (16 MB)
//   X0t transposed  1024*8192            (16 MB)
//   Xc  row-major   1024*8192            (16 MB)   current diffusion state
//   Xct transposed  1024*8192            (16 MB)
//   Ah  f16 supports 2*1024*1024         ( 4 MB)
//   Wt  f16 gathered weights 5*128*128   (160 KB)
// ---------------------------------------------------------------------------
extern "C" void kernel_launch(void* const* d_in, const int* in_sizes, int n_in,
                              void* d_out, int out_size, void* d_ws, size_t ws_size,
                              hipStream_t stream) {
  (void)in_sizes; (void)n_in; (void)out_size; (void)ws_size;
  const float* supports = (const float*)d_in[0];
  const float* inputs   = (const float*)d_in[1];
  const float* state    = (const float*)d_in[2];
  const float* weight   = (const float*)d_in[3];
  const float* biases   = (const float*)d_in[4];
  float* out = (float*)d_out;

  const size_t XN = (size_t)N_NODES * COLS;   // 8M f16 elements
  _Float16* X0  = (_Float16*)d_ws;
  _Float16* X0t = X0  + XN;
  _Float16* Xc  = X0t + XN;
  _Float16* Xct = Xc  + XN;
  _Float16* Ah  = Xct + XN;
  _Float16* Wt  = Ah + (size_t)2 * N_NODES * N_NODES;

  // Stage inputs in f16 (both layouts for x0).
  pack_x0_kernel<<<dim3(COLS / 32, N_NODES / 32), dim3(32, 8), 0, stream>>>(
      inputs, state, X0, X0t);
  convert_a_kernel<<<(2 * N_NODES * N_NODES) / 256, 256, 0, stream>>>(supports, Ah);
  gather_w_kernel<<<(5 * ODIM * FDIM) / 256, 256, 0, stream>>>(weight, Wt);

  // k = 0 term (x0) initializes d_out (+bias).
  out_gemm_kernel<<<(64 * N_NODES) / 64, 256, 0, stream>>>(Wt, X0, biases, out, 1);

  for (int s = 0; s < 2; ++s) {
    const _Float16* As = Ah + (size_t)s * N_NODES * N_NODES;
    // x1 = A @ x0  (write both layouts; transposed copy feeds the next step)
    diff_gemm_kernel<<<dim3(COLS / 256, N_NODES / 64), 256, 0, stream>>>(
        As, X0t, nullptr, Xc, Xct, 1.0f, /*has_z=*/0, /*write_t=*/1);
    out_gemm_kernel<<<(64 * N_NODES) / 64, 256, 0, stream>>>(
        Wt + (size_t)(1 + 2 * s) * ODIM * FDIM, Xc, biases, out, 0);
    // x2 = 2*(A @ x1) - x0  (row-major only; overwrites Xc after it was consumed)
    diff_gemm_kernel<<<dim3(COLS / 256, N_NODES / 64), 256, 0, stream>>>(
        As, Xct, X0t, Xc, nullptr, 2.0f, /*has_z=*/1, /*write_t=*/0);
    out_gemm_kernel<<<(64 * N_NODES) / 64, 256, 0, stream>>>(
        Wt + (size_t)(2 + 2 * s) * ODIM * FDIM, Xc, biases, out, 0);
  }
}
